// LSTMFeatureExtractor_22497038696940
// MI455X (gfx1250) — matrix operand order repaired and checked by reading, not compiled, not run on hardware
//
#include <hip/hip_runtime.h>
#include <hip/hip_bf16.h>
#include <cstdint>
#include <cstddef>

// ---------------------------------------------------------------------------
// 2-layer LSTM (T=1024, B=64, I=256, H=512) for MI455X / gfx1250.
//   Phase A: bf16 WMMA GEMM  xg = X * W_ih^T + (b_ih + b_hh)   (fp32 out)
//   Phase B: persistent scan kernel, 64 WGs, grid barrier per timestep,
//            w_hh resident in LDS, h exchanged via L2 (double buffered).
// LDS staging uses GLOBAL_LOAD_ASYNC_TO_LDS_B128 (ASYNCcnt) when available.
// ---------------------------------------------------------------------------

typedef __attribute__((ext_vector_type(16))) __bf16       v16bf;
typedef __attribute__((ext_vector_type(8)))  float        v8f;
typedef __attribute__((ext_vector_type(4)))  unsigned int u32x4;
typedef __attribute__((ext_vector_type(4)))  int          i32x4;

union V16 { v16bf v; u32x4 q[2]; };   // 16 bf16 = 8 VGPRs = two 16B chunks

#define T_STEPS 1024
#define BATCH   64
#define IN0     256
#define HID     512
#define G4      2048            // 4*H
#define TB      (T_STEPS * BATCH)
#define NWG     64              // persistent workgroups in scan
#define HSLICE  8               // hidden units per WG (HID / NWG)
#define NCOL    32              // gate columns per WG (4 * HSLICE)
#define HP      520             // padded LDS row stride (bf16 elems, 16B mult)

// ---- async global -> LDS (CDNA5), with safe fallback ----------------------
#if defined(__AMDGCN__) && __has_builtin(__builtin_amdgcn_global_load_async_to_lds_b128)
#define HAS_ASYNC_LDS 1
#else
#define HAS_ASYNC_LDS 0
#endif

#if HAS_ASYNC_LDS
__device__ __forceinline__ void async_copy16(void* lds, const void* gl) {
  // signature (from hipcc diagnostic): (v4i* src_global, v4i* dst_lds, imm, imm)
  __builtin_amdgcn_global_load_async_to_lds_b128(
      (i32x4*)gl, (i32x4*)lds, /*offset=*/0, /*cpol=*/0);
}
__device__ __forceinline__ void async_wait0() {
  __builtin_amdgcn_s_wait_asynccnt(0);
}
#else
__device__ __forceinline__ void async_copy16(void* lds, const void* gl) {
  *(u32x4*)lds = *(const u32x4*)gl;
}
__device__ __forceinline__ void async_wait0() {}
#endif

// -------------------------------- utility kernels --------------------------

__global__ __launch_bounds__(256) void f32_to_bf16_kernel(
    const float* __restrict__ src, __bf16* __restrict__ dst, int n) {
  int i = blockIdx.x * 256 + threadIdx.x;
  int stride = gridDim.x * 256;
  for (; i < n; i += stride) dst[i] = (__bf16)src[i];
}

__global__ __launch_bounds__(256) void bias_combine_kernel(
    const float* __restrict__ a, const float* __restrict__ b,
    float* __restrict__ o, int n) {
  int i = blockIdx.x * 256 + threadIdx.x;
  if (i < n) o[i] = a[i] + b[i];
}

__global__ __launch_bounds__(256) void scan_init_kernel(
    __bf16* __restrict__ hbuf, unsigned int* __restrict__ ctr, int n) {
  int i = blockIdx.x * 256 + threadIdx.x;
  if (i < n) hbuf[i] = (__bf16)0.0f;
  if (i == 0) *ctr = 0u;
}

// ------------------------- WMMA GEMM: C = A * W^T + bias -------------------
// A: (M,K) bf16 row-major.  W: (N,K) bf16 row-major (so B = W^T, and the
// WMMA B-operand per lane is two 16B chunks of a W row, k 8h.. and 16+8h..).  C: (M,N) f32.
// Block tile 128x128, 8 waves as 2(M)x4(N), each wave 4x2 16x16 tiles,
// K staged 32-wide, double-buffered LDS filled by async-to-LDS copies.

__global__ __launch_bounds__(256) void wmma_gemm_bias_kernel(
    const __bf16* __restrict__ A, const __bf16* __restrict__ W,
    const float* __restrict__ bias, float* __restrict__ C,
    int M, int N, int K) {
  __shared__ __bf16 sA[2][128 * 40];   // 40 = 32 + 8 pad (keeps 16B align)
  __shared__ __bf16 sB[2][128 * 40];

  const int tid  = threadIdx.x;
  const int lane = tid & 31;
  const int wave = tid >> 5;
  const int m0 = blockIdx.y << 7;
  const int n0 = blockIdx.x << 7;
  const int lrow = tid >> 1;            // 0..127 loader row
  const int lcol = (tid & 1) << 4;      // 0 / 16 loader k-offset
  const int wm = (wave >> 2) << 6;      // 0 / 64
  const int wn = (wave & 3) << 5;       // 0,32,64,96
  const int l16  = lane & 15;
  const int kh8  = (lane >> 4) << 3;    // A-operand lane-half K offset
  const int kh16 = (lane >> 4) << 4;    // lane-half 16-elem K offset (unused)

  v8f acc[4][2];
  for (int mi = 0; mi < 4; ++mi)
    for (int ni = 0; ni < 2; ++ni) {
      float bv = bias[n0 + wn + ni * 16 + l16];   // bias depends on N only
      for (int r = 0; r < 8; ++r) acc[mi][ni][r] = bv;
    }

  auto load_tiles = [&](int kt, int buf) {
    const __bf16* ag = A + (size_t)(m0 + lrow) * K + kt + lcol;
    __bf16* da = &sA[buf][lrow * 40 + lcol];
    async_copy16(da, ag);
    async_copy16(da + 8, ag + 8);
    const __bf16* wp = W + (size_t)(n0 + lrow) * K + kt + lcol;
    __bf16* db = &sB[buf][lrow * 40 + lcol];
    async_copy16(db, wp);
    async_copy16(db + 8, wp + 8);
  };

  load_tiles(0, 0);
  const int nk = K >> 5;
  for (int kk = 0; kk < nk; ++kk) {
    async_wait0();                      // tile kk landed in LDS
    __syncthreads();
    if (kk + 1 < nk) load_tiles((kk + 1) << 5, (kk + 1) & 1);  // overlap
    const __bf16* ab = sA[kk & 1];
    const __bf16* bb = sB[kk & 1];
    V16 av[4], bv[2];
    for (int mi = 0; mi < 4; ++mi) {
      const __bf16* p = ab + (wm + mi * 16 + l16) * 40;
      av[mi].q[0] = *(const u32x4*)(p + kh8);        // K 0-7 / 8-15
      av[mi].q[1] = *(const u32x4*)(p + 16 + kh8);   // K 16-23 / 24-31
    }
    for (int ni = 0; ni < 2; ++ni) {
      const __bf16* p = bb + (wn + ni * 16 + l16) * 40 + kh8;
      bv[ni].q[0] = *(const u32x4*)(p);              // K 0-7 / 8-15
      bv[ni].q[1] = *(const u32x4*)(p + 16);         // K 16-23 / 24-31
    }
    for (int mi = 0; mi < 4; ++mi)
      for (int ni = 0; ni < 2; ++ni)
        acc[mi][ni] = __builtin_amdgcn_wmma_f32_16x16x32_bf16(
            false, av[mi].v, false, bv[ni].v, (short)0, acc[mi][ni],
            false, false);
    __syncthreads();   // all reads of buffer kk done before it is refilled
  }

  for (int mi = 0; mi < 4; ++mi)
    for (int ni = 0; ni < 2; ++ni) {
      int m = m0 + wm + mi * 16 + kh8;   // C layout: lanes>=16 start at M+8
      int n = n0 + wn + ni * 16 + l16;
      float* cp = C + (size_t)m * N + n;
      for (int r = 0; r < 8; ++r) cp[(size_t)r * N] = acc[mi][ni][r];
    }
}

// ------------------------------- persistent scan ----------------------------
// Each of 64 WGs owns hidden units j in [wg*8, wg*8+8): local gate columns
// 0..31 map to global gate column (col>>3)*512 + wg*8 + (col&7)  (i,f,g,o).
// Per step: async-stage h_{t-1} (64x512 bf16) into LDS, seed accumulators
// from xg(t), 16 x v_wmma_f32_16x16x32_bf16 per wave, LSTM pointwise, write
// h to double-buffered global hbuf, device-wide barrier via atomic counter.

__global__ __launch_bounds__(256) void lstm_scan_kernel(
    const float* __restrict__ xg,      // (T,B,4H) fp32
    const __bf16* __restrict__ whh,    // (4H,H) bf16
    __bf16* __restrict__ hbuf,         // (2,B,H) bf16 ping-pong
    __bf16* __restrict__ hseq,         // (T,B,H) bf16 (layer 0) or null
    float* __restrict__ outp,          // (T,B,H) fp32 (layer 1) or null
    unsigned int* __restrict__ ctr) {
  extern __shared__ char smem[];
  __bf16* w_lds = (__bf16*)smem;                              // NCOL x HP
  __bf16* h_lds = (__bf16*)(smem + NCOL * HP * 2);            // BATCH x HP
  float*  gates = (float*)(smem + (NCOL + BATCH) * HP * 2);   // BATCH x NCOL
  float*  cst   = gates + BATCH * NCOL;                       // BATCH x HSLICE

  const int tid  = threadIdx.x;
  const int wg   = blockIdx.x;
  const int lane = tid & 31;
  const int wave = tid >> 5;
  const int mt = wave >> 1;            // batch tile 0..3
  const int nt = wave & 1;             // column tile 0..1
  const int l16  = lane & 15;
  const int kh8  = (lane >> 4) << 3;
  const int kh16 = (lane >> 4) << 4;

  // Gather this WG's w_hh slice into LDS once (resident for all 1024 steps).
  for (int c = tid; c < NCOL * (HID / 8); c += 256) {
    int n  = c >> 6;                   // local column 0..31
    int ko = (c & 63) << 3;            // k offset in elems
    int gr = ((n >> 3) << 9) + wg * HSLICE + (n & 7);   // global gate row
    async_copy16(w_lds + n * HP + ko, whh + (size_t)gr * HID + ko);
  }
  for (int p = tid; p < BATCH * HSLICE; p += 256) cst[p] = 0.0f;

  const int col  = nt * 16 + l16;                            // 0..31
  const int gcol = ((col >> 3) << 9) + wg * HSLICE + (col & 7);

  for (int t = 0; t < T_STEPS; ++t) {
    // h_{t-1} from ping-pong buffer (all-WG writes made visible by barrier)
    const __bf16* hsrc = hbuf + (size_t)(t & 1) * BATCH * HID;
    for (int c = tid; c < BATCH * (HID / 8); c += 256) {
      int b  = c >> 6;
      int ko = (c & 63) << 3;
      async_copy16(h_lds + b * HP + ko, hsrc + b * HID + ko);
    }
    async_wait0();
    __syncthreads();

    // accumulator <- precomputed xg(t) tile
    v8f acc;
    {
      const float* xp = xg + ((size_t)t * BATCH + mt * 16 + kh8) * G4 + gcol;
      for (int r = 0; r < 8; ++r) acc[r] = xp[(size_t)r * G4];
      if (t + 1 < T_STEPS)
        __builtin_prefetch(xp + (size_t)BATCH * G4, 0, 1);   // next step's xg
    }

    // gates += h_{t-1} @ w_hh_slice^T : 16 WMMAs along K=512
    const __bf16* arow = h_lds + (mt * 16 + l16) * HP;
    const __bf16* brow = w_lds + col * HP;
    for (int kk = 0; kk < HID / 32; ++kk) {
      int k0 = kk << 5;
      V16 a, b;
      a.q[0] = *(const u32x4*)(arow + k0 + kh8);
      a.q[1] = *(const u32x4*)(arow + k0 + 16 + kh8);
      b.q[0] = *(const u32x4*)(brow + k0 + kh8);
      b.q[1] = *(const u32x4*)(brow + k0 + 16 + kh8);
      acc = __builtin_amdgcn_wmma_f32_16x16x32_bf16(
          false, a.v, false, b.v, (short)0, acc, false, false);
    }

    // publish gate tile to LDS
    {
      float* gp = gates + (mt * 16 + kh8) * NCOL + col;
      for (int r = 0; r < 8; ++r) gp[r * NCOL] = acc[r];
    }
    __syncthreads();

    // pointwise LSTM cell for this WG's 64x8 slice
    for (int p = tid; p < BATCH * HSLICE; p += 256) {
      int b = p >> 3, jl = p & 7;
      const float* g = gates + b * NCOL;
      float ig = 1.0f / (1.0f + __expf(-g[jl]));
      float fg = 1.0f / (1.0f + __expf(-g[8 + jl]));
      float gg = tanhf(g[16 + jl]);
      float og = 1.0f / (1.0f + __expf(-g[24 + jl]));
      float cv = fg * cst[p] + ig * gg;
      cst[p] = cv;
      float hv = og * tanhf(cv);
      int j = wg * HSLICE + jl;
      hbuf[(size_t)((t + 1) & 1) * BATCH * HID + b * HID + j] = (__bf16)hv;
      if (hseq) hseq[((size_t)t * BATCH + b) * HID + j] = (__bf16)hv;
      else      outp[((size_t)t * BATCH + b) * HID + j] = hv;
    }

    // device-wide barrier: release stores, arrive, spin with acquire
    __threadfence();
    __syncthreads();
    if (tid == 0) {
      __hip_atomic_fetch_add(ctr, 1u, __ATOMIC_RELEASE,
                             __HIP_MEMORY_SCOPE_AGENT);
      unsigned target = (unsigned)(t + 1) * NWG;
      while (__hip_atomic_load(ctr, __ATOMIC_ACQUIRE,
                               __HIP_MEMORY_SCOPE_AGENT) < target)
        __builtin_amdgcn_s_sleep(2);
    }
    __syncthreads();
    __threadfence();   // invalidate stale cached h before next-step load
  }
}

// --------------------------------- launcher --------------------------------

extern "C" void kernel_launch(void* const* d_in, const int* in_sizes, int n_in,
                              void* d_out, int out_size, void* d_ws,
                              size_t ws_size, hipStream_t stream) {
  (void)in_sizes; (void)n_in; (void)out_size; (void)ws_size;
  const float* x    = (const float*)d_in[0];
  const float* wih0 = (const float*)d_in[1];
  const float* whh0 = (const float*)d_in[2];
  const float* bih0 = (const float*)d_in[3];
  const float* bhh0 = (const float*)d_in[4];
  const float* wih1 = (const float*)d_in[5];
  const float* whh1 = (const float*)d_in[6];
  const float* bih1 = (const float*)d_in[7];
  const float* bhh1 = (const float*)d_in[8];
  float* out = (float*)d_out;

  char* ws = (char*)d_ws;
  size_t off = 0;
  auto carve = [&](size_t bytes) -> void* {
    void* p = ws + off;
    off += (bytes + 255) & ~(size_t)255;
    return p;
  };
  __bf16* xbf   = (__bf16*)carve((size_t)TB * IN0 * 2);   //  32 MB
  __bf16* h1seq = (__bf16*)carve((size_t)TB * HID * 2);   //  64 MB
  float*  xgbuf = (float*) carve((size_t)TB * G4 * 4);    // 512 MB
  __bf16* wih0b = (__bf16*)carve((size_t)G4 * IN0 * 2);
  __bf16* whh0b = (__bf16*)carve((size_t)G4 * HID * 2);
  __bf16* wih1b = (__bf16*)carve((size_t)G4 * HID * 2);
  __bf16* whh1b = (__bf16*)carve((size_t)G4 * HID * 2);
  float*  bias0 = (float*) carve((size_t)G4 * 4);
  float*  bias1 = (float*) carve((size_t)G4 * 4);
  __bf16* hbuf  = (__bf16*)carve((size_t)2 * BATCH * HID * 2);
  unsigned int* ctr = (unsigned int*)carve(256);

  // fp32 -> bf16 casts (weights once, x once; h1seq produced as bf16 directly)
  f32_to_bf16_kernel<<<4096, 256, 0, stream>>>(x,    xbf,   TB * IN0);
  f32_to_bf16_kernel<<<2048, 256, 0, stream>>>(wih0, wih0b, G4 * IN0);
  f32_to_bf16_kernel<<<2048, 256, 0, stream>>>(whh0, whh0b, G4 * HID);
  f32_to_bf16_kernel<<<2048, 256, 0, stream>>>(wih1, wih1b, G4 * HID);
  f32_to_bf16_kernel<<<2048, 256, 0, stream>>>(whh1, whh1b, G4 * HID);
  bias_combine_kernel<<<8, 256, 0, stream>>>(bih0, bhh0, bias0, G4);
  bias_combine_kernel<<<8, 256, 0, stream>>>(bih1, bhh1, bias1, G4);

  const size_t smem = (size_t)(NCOL + BATCH) * HP * 2 +
                      (size_t)BATCH * NCOL * 4 + (size_t)BATCH * HSLICE * 4;
  dim3 gemm_grid(G4 / 128, TB / 128);

  // ---- layer 0 ----
  wmma_gemm_bias_kernel<<<gemm_grid, 256, 0, stream>>>(
      xbf, wih0b, bias0, xgbuf, TB, G4, IN0);
  scan_init_kernel<<<(2 * BATCH * HID + 255) / 256, 256, 0, stream>>>(
      hbuf, ctr, 2 * BATCH * HID);
  lstm_scan_kernel<<<NWG, 256, smem, stream>>>(
      xgbuf, whh0b, hbuf, h1seq, nullptr, ctr);

  // ---- layer 1 ----
  wmma_gemm_bias_kernel<<<gemm_grid, 256, 0, stream>>>(
      h1seq, wih1b, bias1, xgbuf, TB, G4, HID);
  scan_init_kernel<<<(2 * BATCH * HID + 255) / 256, 256, 0, stream>>>(
      hbuf, ctr, 2 * BATCH * HID);
  lstm_scan_kernel<<<NWG, 256, smem, stream>>>(
      xgbuf, whh1b, hbuf, nullptr, out, ctr);
}
